// MHDM__970662609222
// MI455X (gfx1250) — matrix operand order repaired and checked by reading, compiled, not run on hardware
//
#include <hip/hip_runtime.h>
#include <hip/hip_bf16.h>

typedef __bf16 bf16;
typedef __attribute__((ext_vector_type(16))) __bf16 v16bf;
typedef __attribute__((ext_vector_type(8)))  __bf16 v8bf;
typedef __attribute__((ext_vector_type(8)))  float  v8f;

#define B_  8
#define N_  768
#define D_  2048
#define H_  4
#define NH  192   // N_/H_

__device__ __forceinline__ bf16 f2bf(float f) { return (bf16)f; }

__device__ __forceinline__ v16bf cat16(v8bf a, v8bf b) {
  v16bf r;
#pragma unroll
  for (int i = 0; i < 8; ++i) { r[i] = a[i]; r[i + 8] = b[i]; }
  return r;
}

// A fragment: 16(M) x 32(K), src row-major (rows m, ld elements).
// lane: m = lane&15, kbase = 8*(lane>>4); elems 0-7 -> K=kb..kb+7, elems 8-15 -> K=kb+16..kb+23
__device__ __forceinline__ v16bf load_a(const bf16* base, int ld, int lane) {
  int m = lane & 15, kb = (lane >> 4) * 8;
  const bf16* p = base + (size_t)m * ld + kb;
  return cat16(*(const v8bf*)p, *(const v8bf*)(p + 16));
}

// B fragment: 32(K) x 16(N) with B[k][n] = src[n*ld + k] (k contiguous in memory).
// lane: n = lane&15, kbase = 8*(lane>>4); elems 0-7 -> K=kb..kb+7, elems 8-15 -> K=kb+16..kb+23
__device__ __forceinline__ v16bf load_b(const bf16* base, int ld, int lane) {
  int n = lane & 15, kb = (lane >> 4) * 8;
  const bf16* p = base + (size_t)n * ld + kb;
  return cat16(*(const v8bf*)p, *(const v8bf*)(p + 16));
}

__device__ __forceinline__ v8f wmma_bf16(v16bf a, v16bf b, v8f c) {
  return __builtin_amdgcn_wmma_f32_16x16x32_bf16(false, a, false, b, (short)0, c,
                                                 false, false);
}

// ---------------------------------------------------------------- transpose x
// x (B,N,D) f32 -> xT (B,D,N) bf16
__global__ void k_transpose(const float* __restrict__ x, bf16* __restrict__ xT) {
  __shared__ float t[32][33];
  int b = blockIdx.z;
  int d0 = blockIdx.x * 32, n0 = blockIdx.y * 32;
  int tx = threadIdx.x, ty = threadIdx.y;              // block (32,8)
  const float* xs = x + (size_t)b * N_ * D_;
#pragma unroll
  for (int i = 0; i < 32; i += 8)
    t[ty + i][tx] = xs[(size_t)(n0 + ty + i) * D_ + d0 + tx];
  __syncthreads();
  bf16* xd = xT + (size_t)b * D_ * N_;
#pragma unroll
  for (int i = 0; i < 32; i += 8)
    xd[(size_t)(d0 + ty + i) * N_ + n0 + tx] = f2bf(t[tx][ty + i]);
}

// ------------------------------------------------------------- weight convert
__global__ void k_cvt(const float* __restrict__ s, bf16* __restrict__ d, int n) {
  for (int i = blockIdx.x * blockDim.x + threadIdx.x; i < n;
       i += gridDim.x * blockDim.x)
    d[i] = f2bf(s[i]);
}

// ----------------------------------------------------------------- projection
// qk[b,d,j] = sum_k xT[b,d,k] * Wqk[j,k]; v likewise.
// QK stored (B,H,D,NH) row-major bf16. V stored transposed: VT (B,H,NH,D) bf16.
__global__ void __launch_bounds__(256) k_proj(const bf16* __restrict__ xT,
                                              const bf16* __restrict__ wqk,
                                              const bf16* __restrict__ wv,
                                              bf16* __restrict__ QK,
                                              bf16* __restrict__ VT) {
  int lane = threadIdx.x & 31;
  int wid = (blockIdx.x * blockDim.x + threadIdx.x) >> 5;
  const int TD = D_ / 16, TC = N_ / 16;
  int ct = wid % TC; int tmp = wid / TC;
  int dt = tmp % TD; int b = tmp / TD;
  int d0 = dt * 16, c0 = ct * 16;
  int h = c0 / NH, cl = c0 % NH;

  const bf16* ab = xT + ((size_t)b * D_ + d0) * N_;
  const bf16* bq = wqk + (size_t)c0 * N_;
  const bf16* bv = wv + (size_t)c0 * N_;
  v8f cq = {}; v8f cv = {};
  for (int k0 = 0; k0 < N_; k0 += 32) {
    v16bf a = load_a(ab + k0, N_, lane);
    cq = wmma_bf16(a, load_b(bq + k0, N_, lane), cq);
    cv = wmma_bf16(a, load_b(bv + k0, N_, lane), cv);
  }
  int hl = lane >> 4, n = lane & 15;
  bf16* qko = QK + ((size_t)b * H_ + h) * D_ * NH;
#pragma unroll
  for (int r = 0; r < 8; ++r)
    qko[(size_t)(d0 + 8 * hl + r) * NH + cl + n] = f2bf(cq[r]);
  v8bf pk;
#pragma unroll
  for (int r = 0; r < 8; ++r) pk[r] = f2bf(cv[r]);
  *(v8bf*)(VT + (((size_t)b * H_ + h) * NH + cl + n) * D_ + d0 + 8 * hl) = pk;
}

// -------------------------------------------------------------------- q2 rows
__global__ void k_q2(const bf16* __restrict__ QK, float* __restrict__ q2) {
  int i = blockIdx.x * blockDim.x + threadIdx.x;   // (b*H+h)*D + d
  const bf16* p = QK + (size_t)i * NH;
  float s = 0.f;
#pragma unroll
  for (int j = 0; j < NH; j += 8) {
    v8bf v = *(const v8bf*)(p + j);
#pragma unroll
    for (int e = 0; e < 8; ++e) { float f = (float)v[e]; s += f * f; }
  }
  q2[i] = s;
}

// ---------------------------------------------------------- flash attention
// One wave owns 16 Q rows of one (b,h); streams over all 2048 keys in blocks
// of 32; online softmax; accumulates out 16x192 in 12 f32 C-tiles.
// Writes W (B, D, N_) bf16 at [b, d, h*NH + c].
__global__ void __launch_bounds__(128) k_attn(const bf16* __restrict__ QK,
                                              const bf16* __restrict__ VT,
                                              const float* __restrict__ q2,
                                              bf16* __restrict__ Wo) {
  __shared__ alignas(16) bf16 smem[4][16 * NH];     // per-wave staging
  int lane = threadIdx.x & 31;
  int w = threadIdx.x >> 5;
  int wid = blockIdx.x * 4 + w;
  const int TD = D_ / 16;
  int dt = wid % TD; int bh = wid / TD;
  int h = bh % H_; int b = bh / H_;
  int row0 = dt * 16;
  const bf16* qkh = QK + (size_t)bh * D_ * NH;
  const bf16* vth = VT + (size_t)bh * NH * D_;
  const float* q2h = q2 + (size_t)bh * D_;
  int hl = lane >> 4, ln = lane & 15;

  v16bf aq[6];                                      // resident Q fragments
#pragma unroll
  for (int t = 0; t < 6; ++t)
    aq[t] = load_a(qkh + (size_t)row0 * NH + t * 32, NH, lane);

  float q2own = q2h[row0 + ln];
  float q2r[8];
#pragma unroll
  for (int r = 0; r < 8; ++r) q2r[r] = __shfl(q2own, r + 8 * hl, 32);

  float mi[8], li[8];
#pragma unroll
  for (int r = 0; r < 8; ++r) { mi[r] = -__builtin_inff(); li[r] = 0.f; }
  v8f acc[12];
#pragma unroll
  for (int c = 0; c < 12; ++c) acc[c] = (v8f){};

  bf16* pl = &smem[w][0];                           // 16x32 P staging

  for (int j0 = 0; j0 < D_; j0 += 32) {
    float lg[2][8];
#pragma unroll
    for (int t = 0; t < 2; ++t) {
      v8f s = {};
#pragma unroll
      for (int kk = 0; kk < 6; ++kk)
        s = wmma_bf16(aq[kk],
                      load_b(qkh + (size_t)(j0 + 16 * t) * NH + kk * 32, NH, lane),
                      s);
      float q2c = q2h[j0 + 16 * t + ln];
#pragma unroll
      for (int r = 0; r < 8; ++r) lg[t][r] = 2.f * s[r] - q2r[r] - q2c;
    }
    float mnew[8], sc[8];
#pragma unroll
    for (int r = 0; r < 8; ++r) {
      float m = fmaxf(lg[0][r], lg[1][r]);
#pragma unroll
      for (int x = 1; x < 16; x <<= 1) m = fmaxf(m, __shfl_xor(m, x, 32));
      mnew[r] = fmaxf(mi[r], m);
      sc[r] = __expf(mi[r] - mnew[r]);
      mi[r] = mnew[r];
    }
#pragma unroll
    for (int c = 0; c < 12; ++c)
#pragma unroll
      for (int r = 0; r < 8; ++r) acc[c][r] *= sc[r];
#pragma unroll
    for (int r = 0; r < 8; ++r) {
      float p0 = __expf(lg[0][r] - mnew[r]);
      float p1 = __expf(lg[1][r] - mnew[r]);
      pl[(8 * hl + r) * 32 + ln]      = f2bf(p0);
      pl[(8 * hl + r) * 32 + 16 + ln] = f2bf(p1);
      float s = p0 + p1;
#pragma unroll
      for (int x = 1; x < 16; x <<= 1) s += __shfl_xor(s, x, 32);
      li[r] = li[r] * sc[r] + s;
    }
    __syncthreads();
    v16bf pa = load_a(pl, 32, lane);                // P as A-fragment (16x32)
    __syncthreads();
#pragma unroll
    for (int c = 0; c < 12; ++c)
      acc[c] = wmma_bf16(pa, load_b(vth + (size_t)(c * 16) * D_ + j0, D_, lane),
                         acc[c]);
  }

#pragma unroll
  for (int r = 0; r < 8; ++r) li[r] = 1.f / li[r];
  bf16* ol = &smem[w][0];                           // 16x192 out staging
#pragma unroll
  for (int c = 0; c < 12; ++c)
#pragma unroll
    for (int r = 0; r < 8; ++r)
      ol[(8 * hl + r) * NH + c * 16 + ln] = f2bf(acc[c][r] * li[r]);
  __syncthreads();
  bf16* wg = Wo + ((size_t)b * D_ + row0) * N_ + h * NH;
  int cs = hl * 96;
#pragma unroll
  for (int i = 0; i < 12; ++i)
    *(v8bf*)(wg + (size_t)ln * N_ + cs + i * 8) =
        *(v8bf*)(ol + ln * NH + cs + i * 8);
}

// -------------------------------------------------------- output GEMM + store
// fo[b,d,j] = sum_c W[b,d,c] * Wout[j,c]; write out[b,j,d] (B,N,D) f32.
__global__ void __launch_bounds__(256) k_out(const bf16* __restrict__ Wbuf,
                                             const bf16* __restrict__ wout,
                                             float* __restrict__ out) {
  int lane = threadIdx.x & 31;
  int wid = (blockIdx.x * blockDim.x + threadIdx.x) >> 5;
  const int TD = D_ / 16, TJ = N_ / 16;
  int jt = wid % TJ; int tmp = wid / TJ;
  int dt = tmp % TD; int b = tmp / TD;
  int d0 = dt * 16, j0 = jt * 16;
  const bf16* ab = Wbuf + ((size_t)b * D_ + d0) * N_;
  const bf16* bb = wout + (size_t)j0 * N_;
  v8f c = {};
  for (int k0 = 0; k0 < N_; k0 += 32)
    c = wmma_bf16(load_a(ab + k0, N_, lane), load_b(bb + k0, N_, lane), c);
  int hl = lane >> 4, n = lane & 15;
  float* p = out + ((size_t)b * N_ + j0 + n) * D_ + d0 + 8 * hl;
  float4 s0 = {c[0], c[1], c[2], c[3]};
  float4 s1 = {c[4], c[5], c[6], c[7]};
  *(float4*)p = s0;
  *(float4*)(p + 4) = s1;
}

extern "C" void kernel_launch(void* const* d_in, const int* in_sizes, int n_in,
                              void* d_out, int out_size, void* d_ws, size_t ws_size,
                              hipStream_t stream) {
  const float* x    = (const float*)d_in[0];
  const float* Wqk  = (const float*)d_in[1];
  const float* Wv   = (const float*)d_in[2];
  const float* Wout = (const float*)d_in[3];
  float* out = (float*)d_out;

  char* ws = (char*)d_ws;
  size_t oXT = 0;
  size_t oQK = oXT + (size_t)B_ * D_ * N_ * 2;
  size_t oVT = oQK + (size_t)B_ * H_ * D_ * NH * 2;
  size_t oW  = oVT + (size_t)B_ * H_ * NH * D_ * 2;
  size_t oQ2 = oW  + (size_t)B_ * D_ * N_ * 2;
  size_t oWq = oQ2 + (size_t)B_ * H_ * D_ * 4;
  size_t oWv = oWq + (size_t)N_ * N_ * 2;
  size_t oWo = oWv + (size_t)N_ * N_ * 2;

  bf16* xT   = (bf16*)(ws + oXT);
  bf16* QK   = (bf16*)(ws + oQK);
  bf16* VT   = (bf16*)(ws + oVT);
  bf16* Wbuf = (bf16*)(ws + oW);
  float* q2  = (float*)(ws + oQ2);
  bf16* Wqb  = (bf16*)(ws + oWq);
  bf16* Wvb  = (bf16*)(ws + oWv);
  bf16* Wob  = (bf16*)(ws + oWo);

  k_transpose<<<dim3(D_ / 32, N_ / 32, B_), dim3(32, 8), 0, stream>>>(x, xT);
  k_cvt<<<2304, 256, 0, stream>>>(Wqk, Wqb, N_ * N_);
  k_cvt<<<2304, 256, 0, stream>>>(Wv, Wvb, N_ * N_);
  k_cvt<<<2304, 256, 0, stream>>>(Wout, Wob, N_ * N_);

  // 8 * 128 * 48 = 49152 tiles, 8 waves/block
  k_proj<<<6144, 256, 0, stream>>>(xT, Wqb, Wvb, QK, VT);
  k_q2<<<(B_ * H_ * D_) / 256, 256, 0, stream>>>(QK, q2);

  // 8 * 4 * 128 = 4096 waves, 4 waves/block
  k_attn<<<1024, 128, 0, stream>>>(QK, VT, q2, Wbuf);

  k_out<<<6144, 256, 0, stream>>>(Wbuf, Wob, out);
}
